// Decoder_63453846831351
// MI455X (gfx1250) — matrix operand order repaired and checked by reading, not compiled, not run on hardware
//
#include <hip/hip_runtime.h>
#include <hip/hip_bf16.h>

// ---------------------------------------------------------------------------
// CDNA5 WMMA (wave32): v_wmma_f32_16x16x32_bf16
// ---------------------------------------------------------------------------
typedef __attribute__((ext_vector_type(16))) __bf16 bf16x16;
typedef __attribute__((ext_vector_type(8)))  float  floatx8;

#define BB   64
#define SS   128
#define EE   512
#define HH   512
#define H2   1024
#define VV   32000

__device__ __forceinline__ float sigmoidf_(float x) {
  return 1.0f / (1.0f + __expf(-x));
}

// 16x32 bf16 A-tile from row-major f32. CDNA5 layout:
// lanes 0-15: M=lane, K {0..7,16..23}; lanes 16-31: M=lane-16, K {8..15,24..31}
// ka = (lane>>4)*8. All addresses 16B aligned -> b128 loads.
__device__ __forceinline__ bf16x16 load_a_f32_bf16(const float* __restrict__ Arow,
                                                   int k0, int ka) {
  const float4 x0 = *(const float4*)(Arow + k0 + ka);
  const float4 x1 = *(const float4*)(Arow + k0 + ka + 4);
  const float4 y0 = *(const float4*)(Arow + k0 + ka + 16);
  const float4 y1 = *(const float4*)(Arow + k0 + ka + 20);
  bf16x16 a;
  a[0]=(__bf16)x0.x; a[1]=(__bf16)x0.y; a[2]=(__bf16)x0.z; a[3]=(__bf16)x0.w;
  a[4]=(__bf16)x1.x; a[5]=(__bf16)x1.y; a[6]=(__bf16)x1.z; a[7]=(__bf16)x1.w;
  a[8]=(__bf16)y0.x; a[9]=(__bf16)y0.y; a[10]=(__bf16)y0.z; a[11]=(__bf16)y0.w;
  a[12]=(__bf16)y1.x; a[13]=(__bf16)y1.y; a[14]=(__bf16)y1.z; a[15]=(__bf16)y1.w;
  return a;
}

// 32x16 bf16 B-tile from row-major W[N,K] (A @ W^T):
// lanes 0-15: col=lane, K {0..7,16..23} ; lanes 16-31: col=lane-16, K {8..15,24..31}
// kb = (lane>>4)*8 -> 4x b128 loads.
__device__ __forceinline__ bf16x16 load_b_f32_bf16(const float* __restrict__ Wrow,
                                                   int k0, int kb) {
  const float4 x0 = *(const float4*)(Wrow + k0 + kb);
  const float4 x1 = *(const float4*)(Wrow + k0 + kb + 4);
  const float4 x2 = *(const float4*)(Wrow + k0 + kb + 16);
  const float4 x3 = *(const float4*)(Wrow + k0 + kb + 20);
  bf16x16 b;
  b[0]=(__bf16)x0.x; b[1]=(__bf16)x0.y; b[2]=(__bf16)x0.z; b[3]=(__bf16)x0.w;
  b[4]=(__bf16)x1.x; b[5]=(__bf16)x1.y; b[6]=(__bf16)x1.z; b[7]=(__bf16)x1.w;
  b[8]=(__bf16)x2.x; b[9]=(__bf16)x2.y; b[10]=(__bf16)x2.z; b[11]=(__bf16)x2.w;
  b[12]=(__bf16)x3.x; b[13]=(__bf16)x3.y; b[14]=(__bf16)x3.z; b[15]=(__bf16)x3.w;
  return b;
}

// ---------------------------------------------------------------------------
// C[64,N] (+)= A[64,K] @ W[N,K]^T + bias0 + bias1
// Block: 256 thr = 8 waves; block covers FULL M=64 x 128 cols.
// Each wave owns one 16-col strip and 4 M-tiles (4x v8f accumulators), so
// every streamed weight tile (the HBM-bound term) is loaded exactly once
// and reused by 4 WMMAs. gridDim.x = N/128. K % 32 == 0.
// ---------------------------------------------------------------------------
__global__ void wmma_gemm_nt(const float* __restrict__ A, int lda,
                             const float* __restrict__ W, int ldw,
                             const float* __restrict__ bias0,
                             const float* __restrict__ bias1,
                             float* __restrict__ C, int ldc,
                             int K, int accumulate) {
  const int lane = threadIdx.x & 31;
  const int wave = threadIdx.x >> 5;
  const int col0 = blockIdx.x * 128 + wave * 16;
  const int ka = (lane >> 4) * 8;
  const int kb = (lane >> 4) * 8;

  floatx8 acc[4];
#pragma unroll
  for (int m = 0; m < 4; ++m)
#pragma unroll
    for (int j = 0; j < 8; ++j) acc[m][j] = 0.0f;

  const float* Arow = A + (long)(lane & 15) * lda;          // + m*16*lda per tile
  const float* Wrow = W + (long)(col0 + (lane & 15)) * ldw;

  for (int k0 = 0; k0 < K; k0 += 32) {
    // keep the HBM weight stream primed (global_prefetch_b8)
    __builtin_prefetch(Wrow + k0 + 512, 0, 1);
    bf16x16 b = load_b_f32_bf16(Wrow, k0, kb);
#pragma unroll
    for (int m = 0; m < 4; ++m) {
      bf16x16 a = load_a_f32_bf16(Arow + (long)m * 16 * lda, k0, ka);
      acc[m] = __builtin_amdgcn_wmma_f32_16x16x32_bf16(
          false, a, false, b, (short)0, acc[m], false, false);
    }
  }

  const int col = col0 + (lane & 15);
  float badd = 0.0f;
  if (bias0) badd += bias0[col];
  if (bias1) badd += bias1[col];
#pragma unroll
  for (int m = 0; m < 4; ++m) {
    const int rbase = m * 16 + (lane >> 4) * 8;
#pragma unroll
    for (int j = 0; j < 8; ++j) {
      long idx = (long)(rbase + j) * ldc + col;
      float v = acc[m][j] + badd;
      if (accumulate) v += C[idx];
      C[idx] = v;
    }
  }
}

// ---------------------------------------------------------------------------
// Embedding gather: x0[:,0:512], xfc[:,2048:2560]
// ---------------------------------------------------------------------------
__global__ void embed_kernel(const int* __restrict__ tok,
                             const float* __restrict__ emb,
                             float* __restrict__ x0, float* __restrict__ xfc) {
  int idx = blockIdx.x * 256 + threadIdx.x;      // 64*512
  int b = idx >> 9, e = idx & 511;
  float v = emb[(long)tok[b] * EE + e];
  x0[b * 1536 + e]         = v;
  xfc[b * 2560 + 2048 + e] = v;
}

// ---------------------------------------------------------------------------
// Fused attention energy: 16 enc rows per block,
//   logits[row] = sum_n v[n] * tanh( q[b,n] + enc_row . Wenc[n,:] )
// 8 waves x 4 col-tiles, K=1024 bf16 WMMA; tanh + v-dot fused via ds_add_f32.
// grid = 512 blocks.
// ---------------------------------------------------------------------------
__global__ void attn_energy_kernel(const float* __restrict__ enc,    // [8192,1024]
                                   const float* __restrict__ attn_w, // [512,2048]
                                   const float* __restrict__ q,      // [64,512]
                                   const float* __restrict__ v_w,    // [512]
                                   float* __restrict__ logits) {     // [64*128]
  __shared__ float red[16];
  const int lane = threadIdx.x & 31;
  const int wave = threadIdx.x >> 5;
  const int row0 = blockIdx.x * 16;
  const int b    = row0 >> 7;
  const int ka = (lane >> 4) * 8;
  const int kb = (lane >> 4) * 8;

  if (threadIdx.x < 16) red[threadIdx.x] = 0.0f;
  __syncthreads();

  floatx8 acc[4];
#pragma unroll
  for (int t = 0; t < 4; ++t)
#pragma unroll
    for (int j = 0; j < 8; ++j) acc[t][j] = 0.0f;

  const float* Arow = enc + (long)(row0 + (lane & 15)) * H2;
  const float* W0   = attn_w + H2;   // encoder half of attn_w columns

  for (int k0 = 0; k0 < H2; k0 += 32) {
    bf16x16 a = load_a_f32_bf16(Arow, k0, ka);
#pragma unroll
    for (int t = 0; t < 4; ++t) {
      const int c0 = wave * 64 + t * 16;
      const float* Wrow = W0 + (long)(c0 + (lane & 15)) * 2048;
      bf16x16 bt = load_b_f32_bf16(Wrow, k0, kb);
      acc[t] = __builtin_amdgcn_wmma_f32_16x16x32_bf16(
          false, a, false, bt, (short)0, acc[t], false, false);
    }
  }

  const int rb = (lane >> 4) * 8;
  float part[8];
#pragma unroll
  for (int j = 0; j < 8; ++j) part[j] = 0.0f;
#pragma unroll
  for (int t = 0; t < 4; ++t) {
    const int col = wave * 64 + t * 16 + (lane & 15);
    const float vw = v_w[col];
    const float qv = q[b * HH + col];
#pragma unroll
    for (int j = 0; j < 8; ++j) part[j] += vw * tanhf(qv + acc[t][j]);
  }
#pragma unroll
  for (int j = 0; j < 8; ++j) atomicAdd(&red[rb + j], part[j]);  // ds_add_f32
  __syncthreads();

  if (threadIdx.x < 16) logits[row0 + threadIdx.x] = red[threadIdx.x];
}

// ---------------------------------------------------------------------------
// Masked softmax over S=128 + context; block per b.
// ---------------------------------------------------------------------------
__global__ void softmax_context_kernel(const float* __restrict__ logits,
                                       const int* __restrict__ mask,
                                       const float* __restrict__ enc,
                                       float* __restrict__ attnw_out,
                                       float* __restrict__ xfc,
                                       float* __restrict__ x0) {
  __shared__ float w[SS];
  __shared__ float sred[2];
  const int b = blockIdx.x, tid = threadIdx.x;

  if (tid < SS) {
    float l = logits[b * SS + tid];
    if (mask[b * SS + tid] == 0) l = -1e10f;
    w[tid] = l;
  }
  __syncthreads();
  if (tid == 0) {
    float mx = w[0];
    for (int s = 1; s < SS; ++s) mx = fmaxf(mx, w[s]);
    sred[0] = mx;
  }
  __syncthreads();
  if (tid < SS) w[tid] = __expf(w[tid] - sred[0]);
  __syncthreads();
  if (tid == 0) {
    float s = 0.0f;
    for (int i = 0; i < SS; ++i) s += w[i];
    sred[1] = s;
  }
  __syncthreads();
  if (tid < SS) {
    float wt = w[tid] / sred[1];
    w[tid] = wt;
    attnw_out[b * SS + tid] = wt;
  }
  __syncthreads();

  for (int d = tid; d < H2; d += 256) {
    float accv = 0.0f;
    const float* e = enc + (long)b * SS * H2 + d;
    for (int s = 0; s < SS; ++s) accv += w[s] * e[s * H2];
    xfc[b * 2560 + 1024 + d] = accv;
    x0[b * 1536 + 512 + d]   = accv;
  }
}

// ---------------------------------------------------------------------------
// LSTM pointwise gates (i,f,g,o), c/h update
// ---------------------------------------------------------------------------
__global__ void lstm_act_kernel(const float* __restrict__ gates,   // [64,4096]
                                const float* __restrict__ c_prev,  // [64,1024]
                                float* __restrict__ h_out,
                                float* __restrict__ c_out,
                                float* __restrict__ h_out2, int ld2) {
  int idx = blockIdx.x * 256 + threadIdx.x;     // 64*1024
  int b = idx >> 10, d = idx & 1023;
  const float* g = gates + (long)b * 4096;
  float i  = sigmoidf_(g[d]);
  float f  = sigmoidf_(g[1024 + d]);
  float gg = tanhf(g[2048 + d]);
  float o  = sigmoidf_(g[3072 + d]);
  float c  = f * c_prev[idx] + i * gg;
  float h  = o * tanhf(c);
  h_out[idx] = h;
  c_out[idx] = c;
  if (h_out2) h_out2[(long)b * ld2 + d] = h;
}

// ---------------------------------------------------------------------------
extern "C" void kernel_launch(void* const* d_in, const int* in_sizes, int n_in,
                              void* d_out, int out_size, void* d_ws, size_t ws_size,
                              hipStream_t stream) {
  (void)in_sizes; (void)n_in; (void)out_size; (void)ws_size;

  const int*   tok    = (const int*)  d_in[0];
  const float* hidden = (const float*)d_in[1];   // [2,64,1024]
  const float* cell   = (const float*)d_in[2];   // [2,64,1024]
  const float* enc    = (const float*)d_in[3];   // [64,128,1024]
  const int*   mask   = (const int*)  d_in[4];   // [64,128]
  const float* emb    = (const float*)d_in[5];   // [32000,512]
  const float* attn_w = (const float*)d_in[6];   // [512,2048]
  const float* attn_b = (const float*)d_in[7];
  const float* v_w    = (const float*)d_in[8];
  const float* w_ih0  = (const float*)d_in[9];   // [4096,1536]
  const float* w_hh0  = (const float*)d_in[10];  // [4096,1024]
  const float* b_ih0  = (const float*)d_in[11];
  const float* b_hh0  = (const float*)d_in[12];
  const float* w_ih1  = (const float*)d_in[13];  // [4096,1024]
  const float* w_hh1  = (const float*)d_in[14];  // [4096,1024]
  const float* b_ih1  = (const float*)d_in[15];
  const float* b_hh1  = (const float*)d_in[16];
  const float* fc_w   = (const float*)d_in[17];  // [32000,2560]
  const float* fc_b   = (const float*)d_in[18];

  float* out = (float*)d_out;
  float* ws  = (float*)d_ws;

  // workspace layout (floats)
  float* x0     = ws;               // [64,1536]  (embedded|context)
  float* xfc    = x0 + 98304;       // [64,2560]  (h1|context|embedded)
  float* q      = xfc + 163840;     // [64,512]
  float* logits = q + 32768;        // [64,128]
  float* gates0 = logits + 8192;    // [64,4096]
  float* gates1 = gates0 + 262144;  // [64,4096]

  // output layout (return order): prediction, new_hidden, new_cell, attn_weights
  float* pred  = out;                       // 64*32000
  float* h_new = out + 2048000;             // [2,64,1024]
  float* c_new = h_new + 131072;            // [2,64,1024]
  float* attnw = c_new + 131072;            // [64,128]

  const float* h_top = hidden + 65536;      // hidden[-1] == hidden[1]

  // 1) embedding gather
  embed_kernel<<<128, 256, 0, stream>>>(tok, emb, x0, xfc);

  // 2) q = h_top @ attn_w[:, :1024]^T + attn_b        [64,512]
  wmma_gemm_nt<<<4, 256, 0, stream>>>(
      h_top, 1024, attn_w, 2048, attn_b, nullptr, q, 512, 1024, 0);

  // 3) fused energy GEMM + tanh + v-dot -> logits
  attn_energy_kernel<<<512, 256, 0, stream>>>(enc, attn_w, q, v_w, logits);

  // 4) masked softmax + context
  softmax_context_kernel<<<64, 256, 0, stream>>>(logits, mask, enc, attnw, xfc, x0);

  // 5) LSTM layer 0
  wmma_gemm_nt<<<32, 256, 0, stream>>>(
      x0, 1536, w_ih0, 1536, b_ih0, b_hh0, gates0, 4096, 1536, 0);
  wmma_gemm_nt<<<32, 256, 0, stream>>>(
      hidden, 1024, w_hh0, 1024, nullptr, nullptr, gates0, 4096, 1024, 1);
  lstm_act_kernel<<<256, 256, 0, stream>>>(gates0, cell, h_new, c_new, nullptr, 0);

  // 6) LSTM layer 1
  wmma_gemm_nt<<<32, 256, 0, stream>>>(
      h_new, 1024, w_ih1, 1024, b_ih1, b_hh1, gates1, 4096, 1024, 0);
  wmma_gemm_nt<<<32, 256, 0, stream>>>(
      h_top, 1024, w_hh1, 1024, nullptr, nullptr, gates1, 4096, 1024, 1);
  lstm_act_kernel<<<256, 256, 0, stream>>>(gates1, cell + 65536,
                                           h_new + 65536, c_new + 65536,
                                           xfc, 2560);   // h1 -> xfc[:, :1024]

  // 7) vocab projection: weights streamed once (328 MB -> ~14 us HBM floor)
  wmma_gemm_nt<<<250, 256, 0, stream>>>(
      xfc, 2560, fc_w, 2560, fc_b, nullptr, pred, 32000, 2560, 0);
}
